// GNN_69922067578969
// MI455X (gfx1250) — compile-verified, matrix operand order checked by reading, not run on hardware
//
#include <hip/hip_runtime.h>
#include <hip/hip_bf16.h>

typedef __attribute__((ext_vector_type(16))) _Float16 v16h;
typedef __attribute__((ext_vector_type(8)))  float    v8f;

#define GN_N   50000
#define GN_E   800000
#define GN_G   500
#define GN_H   64
#define GN_L   4
#define GN_OUT 10
#define GN_EPS 1e-5f

__device__ __forceinline__ v8f wmma_f16(v16h a, v16h b, v8f c) {
  return __builtin_amdgcn_wmma_f32_16x16x32_f16(false, a, false, b, (short)0, c,
                                                false, false);
}

// Load 16 consecutive f32 from a (64B-aligned) row, convert to f16 vector.
__device__ __forceinline__ v16h load16_f32_to_f16(const float* __restrict__ p) {
  const float4* q = (const float4*)p;
  v16h r;
#pragma unroll
  for (int i = 0; i < 4; ++i) {
    float4 v = q[i];
    r[4 * i + 0] = (_Float16)v.x;
    r[4 * i + 1] = (_Float16)v.y;
    r[4 * i + 2] = (_Float16)v.z;
    r[4 * i + 3] = (_Float16)v.w;
  }
  return r;
}

// ---- Pack a row-major f32 [K x 64] weight matrix into WMMA B fragments. ----
// Fragment f = c*4 + t (c = K-chunk of 32, t = N-tile of 16).
// Within a fragment: lane L holds column n = t*16 + L%16, K values
// c*32 + (L/16)*16 + j  (j = 0..15) as 16 consecutive f16.
__global__ void pack_w_kernel(const float* __restrict__ W,
                              _Float16* __restrict__ out, int K) {
  int id = blockIdx.x * blockDim.x + threadIdx.x;
  if (id >= K * 64) return;
  int j    = id & 15;
  int lane = (id >> 4) & 31;
  int f    = id >> 9;
  int c = f >> 2, t = f & 3;
  int k = c * 32 + (lane >> 4) * 16 + j;
  int n = t * 16 + (lane & 15);
  out[id] = (_Float16)W[k * 64 + n];
}

// ---- degree ----
__global__ void deg_kernel(const int* __restrict__ dst, float* __restrict__ deg) {
  int e = blockIdx.x * blockDim.x + threadIdx.x;
  if (e >= GN_E) return;
  atomicAdd(&deg[dst[e]], 1.0f);
}
__global__ void invdeg_kernel(float* __restrict__ deg) {
  int n = blockIdx.x * blockDim.x + threadIdx.x;
  if (n >= GN_N) return;
  deg[n] = 1.0f / fmaxf(deg[n], 1.0f);
}

// ---- message MLP: msg = relu([h_src|h_dst] @ Wm + bm); agg += msg/deg ----
__global__ void msg_kernel(const float* __restrict__ h,
                           const int* __restrict__ src,
                           const int* __restrict__ dst,
                           const _Float16* __restrict__ Bp,
                           const float* __restrict__ bias,
                           const float* __restrict__ invdeg,
                           float* __restrict__ agg) {
  int tile = (blockIdx.x * blockDim.x + threadIdx.x) >> 5;
  if (tile >= GN_E / 16) return;  // wave-uniform
  int lane = threadIdx.x & 31;
  int e0 = tile * 16;
  int m = lane & 15, kh = lane >> 4;

  v8f acc[4] = {v8f{}, v8f{}, v8f{}, v8f{}};
#pragma unroll
  for (int c = 0; c < 4; ++c) {
    int kstart = c * 32 + kh * 16;
    const int* idx = (kstart < 64) ? src : dst;
    int col  = kstart & 63;
    int node = idx[e0 + m];
    v16h a = load16_f32_to_f16(h + (size_t)node * GN_H + col);
#pragma unroll
    for (int t = 0; t < 4; ++t) {
      v16h b = *(const v16h*)(Bp + ((c * 4 + t) * 32 + lane) * 16);
      acc[t] = wmma_f16(a, b, acc[t]);
    }
  }

  int n_local = lane & 15;
  int mbase = (lane >> 4) * 8;
  float bb[4];
#pragma unroll
  for (int t = 0; t < 4; ++t) bb[t] = bias[t * 16 + n_local];
#pragma unroll
  for (int r = 0; r < 8; ++r) {
    int d = dst[e0 + mbase + r];
    float w = invdeg[d];
    float* aggrow = agg + (size_t)d * GN_H;
#pragma unroll
    for (int t = 0; t < 4; ++t) {
      float v = acc[t][r] + bb[t];
      v = v > 0.0f ? v : 0.0f;
      atomicAdd(aggrow + t * 16 + n_local, v * w);
    }
  }
}

// ---- node update: u = PReLU([h|agg] @ Wu + bu); accumulate BN stats ----
__global__ void update_kernel(const float* __restrict__ h,
                              const float* __restrict__ agg,
                              const _Float16* __restrict__ Bp,
                              const float* __restrict__ bias,
                              const float* __restrict__ slope_arr, int layer,
                              float* __restrict__ ubuf,
                              float* __restrict__ stats) {
  int tile = (blockIdx.x * blockDim.x + threadIdx.x) >> 5;
  if (tile >= GN_N / 16) return;  // wave-uniform
  int lane = threadIdx.x & 31;
  int n0 = tile * 16;
  int m = lane & 15, kh = lane >> 4;

  v8f acc[4] = {v8f{}, v8f{}, v8f{}, v8f{}};
#pragma unroll
  for (int c = 0; c < 4; ++c) {
    int kstart = c * 32 + kh * 16;
    const float* base = (kstart < 64) ? h : agg;
    int col = kstart & 63;
    v16h a = load16_f32_to_f16(base + (size_t)(n0 + m) * GN_H + col);
#pragma unroll
    for (int t = 0; t < 4; ++t) {
      v16h b = *(const v16h*)(Bp + ((c * 4 + t) * 32 + lane) * 16);
      acc[t] = wmma_f16(a, b, acc[t]);
    }
  }

  float slope = slope_arr[layer];
  int n_local = lane & 15;
  int mbase = (lane >> 4) * 8;
#pragma unroll
  for (int t = 0; t < 4; ++t) {
    float bb = bias[t * 16 + n_local];
    float s = 0.0f, s2 = 0.0f;
#pragma unroll
    for (int r = 0; r < 8; ++r) {
      float v = acc[t][r] + bb;
      v = v >= 0.0f ? v : slope * v;
      ubuf[(size_t)(n0 + mbase + r) * GN_H + t * 16 + n_local] = v;
      s += v;
      s2 += v * v;
    }
    s  += __shfl_xor(s, 16, 32);
    s2 += __shfl_xor(s2, 16, 32);
    if (lane < 16) {
      atomicAdd(&stats[t * 16 + n_local], s);
      atomicAdd(&stats[64 + t * 16 + n_local], s2);
    }
  }
}

// ---- batchnorm normalize (in place) ----
__global__ void bn_kernel(float* __restrict__ u, const float* __restrict__ stats,
                          const float* __restrict__ gamma,
                          const float* __restrict__ beta) {
  int id = blockIdx.x * blockDim.x + threadIdx.x;
  if (id >= GN_N * GN_H) return;
  int col = id & 63;
  float mu  = stats[col] * (1.0f / GN_N);
  float var = stats[64 + col] * (1.0f / GN_N) - mu * mu;
  float inv = rsqrtf(var + GN_EPS);
  u[id] = (u[id] - mu) * inv * gamma[col] + beta[col];
}

// ---- global mean pool ----
__global__ void cnt_kernel(const int* __restrict__ batch, float* __restrict__ cnt) {
  int n = blockIdx.x * blockDim.x + threadIdx.x;
  if (n >= GN_N) return;
  atomicAdd(&cnt[batch[n]], 1.0f);
}
__global__ void pool_kernel(const float* __restrict__ h,
                            const int* __restrict__ batch,
                            float* __restrict__ g) {
  int id = blockIdx.x * blockDim.x + threadIdx.x;
  if (id >= GN_N * GN_H) return;
  int n = id >> 6, c = id & 63;
  atomicAdd(&g[(size_t)batch[n] * GN_H + c], h[id]);
}
__global__ void gdiv_kernel(float* __restrict__ g, const float* __restrict__ cnt) {
  int id = blockIdx.x * blockDim.x + threadIdx.x;
  if (id >= GN_G * GN_H) return;
  g[id] *= 1.0f / fmaxf(cnt[id >> 6], 1.0f);
}

// ---- MLP head layer 1 via WMMA: t1 = relu(g @ W1 + b1) ----
__global__ void head_kernel(const float* __restrict__ g,
                            const _Float16* __restrict__ Bp,
                            const float* __restrict__ b1,
                            float* __restrict__ t1) {
  int tile = (blockIdx.x * blockDim.x + threadIdx.x) >> 5;
  if (tile >= (GN_G + 15) / 16) return;  // wave-uniform
  int lane = threadIdx.x & 31;
  int m = lane & 15, kh = lane >> 4;
  int row = tile * 16 + m;
  int rowc = row < GN_G ? row : GN_G - 1;

  v8f acc[4] = {v8f{}, v8f{}, v8f{}, v8f{}};
#pragma unroll
  for (int c = 0; c < 2; ++c) {
    int kstart = c * 32 + kh * 16;
    v16h a = load16_f32_to_f16(g + (size_t)rowc * GN_H + kstart);
#pragma unroll
    for (int t = 0; t < 4; ++t) {
      v16h b = *(const v16h*)(Bp + ((c * 4 + t) * 32 + lane) * 16);
      acc[t] = wmma_f16(a, b, acc[t]);
    }
  }

  int n_local = lane & 15;
  int mbase = (lane >> 4) * 8;
#pragma unroll
  for (int t = 0; t < 4; ++t) {
    float bb = b1[t * 16 + n_local];
#pragma unroll
    for (int r = 0; r < 8; ++r) {
      int gr = tile * 16 + mbase + r;
      if (gr < GN_G) {
        float v = acc[t][r] + bb;
        t1[(size_t)gr * GN_H + t * 16 + n_local] = v > 0.0f ? v : 0.0f;
      }
    }
  }
}

// ---- MLP head layer 2 (64 -> 10, tiny) ----
__global__ void head_out_kernel(const float* __restrict__ t1,
                                const float* __restrict__ W2,
                                const float* __restrict__ b2,
                                float* __restrict__ out) {
  int id = blockIdx.x * blockDim.x + threadIdx.x;
  if (id >= GN_G * GN_OUT) return;
  int gi = id / GN_OUT, o = id % GN_OUT;
  float s = b2[o];
#pragma unroll 8
  for (int c = 0; c < GN_H; ++c) s += t1[(size_t)gi * GN_H + c] * W2[c * GN_OUT + o];
  out[id] = s;
}

extern "C" void kernel_launch(void* const* d_in, const int* in_sizes, int n_in,
                              void* d_out, int out_size, void* d_ws, size_t ws_size,
                              hipStream_t stream) {
  const float* x     = (const float*)d_in[0];
  const int*   ei    = (const int*)d_in[1];
  const int*   batch = (const int*)d_in[2];
  const float* Wm    = (const float*)d_in[3];
  const float* bm    = (const float*)d_in[4];
  const float* Wu    = (const float*)d_in[5];
  const float* bu    = (const float*)d_in[6];
  const float* a     = (const float*)d_in[7];
  const float* gamma = (const float*)d_in[8];
  const float* beta  = (const float*)d_in[9];
  const float* W1    = (const float*)d_in[10];
  const float* b1    = (const float*)d_in[11];
  const float* W2    = (const float*)d_in[12];
  const float* b2    = (const float*)d_in[13];

  const int* srcp = ei;
  const int* dstp = ei + GN_E;

  char* base = (char*)d_ws;
  size_t off = 0;
  auto alloc = [&](size_t bytes) -> void* {
    void* p = base + off;
    off += (bytes + 255) & ~(size_t)255;
    return p;
  };
  float* hA     = (float*)alloc((size_t)GN_N * GN_H * 4);
  float* hB     = (float*)alloc((size_t)GN_N * GN_H * 4);
  float* agg    = (float*)alloc((size_t)GN_N * GN_H * 4);
  float* invdeg = (float*)alloc((size_t)GN_N * 4);
  float* stats  = (float*)alloc(128 * 4);
  float* gpool  = (float*)alloc((size_t)GN_G * GN_H * 4);
  float* gcnt   = (float*)alloc((size_t)GN_G * 4);
  float* t1     = (float*)alloc((size_t)GN_G * GN_H * 4);
  _Float16* packWm = (_Float16*)alloc((size_t)GN_L * 128 * GN_H * 2);
  _Float16* packWu = (_Float16*)alloc((size_t)GN_L * 128 * GN_H * 2);
  _Float16* packW1 = (_Float16*)alloc((size_t)GN_H * GN_H * 2);
  (void)ws_size; (void)n_in; (void)in_sizes; (void)out_size;

  // degrees
  hipMemsetAsync(invdeg, 0, (size_t)GN_N * 4, stream);
  deg_kernel<<<(GN_E + 255) / 256, 256, 0, stream>>>(dstp, invdeg);
  invdeg_kernel<<<(GN_N + 255) / 256, 256, 0, stream>>>(invdeg);

  // pack all weights into WMMA B-fragment layout
  for (int l = 0; l < GN_L; ++l) {
    pack_w_kernel<<<(128 * 64 + 255) / 256, 256, 0, stream>>>(
        Wm + (size_t)l * 128 * GN_H, packWm + (size_t)l * 128 * GN_H, 128);
    pack_w_kernel<<<(128 * 64 + 255) / 256, 256, 0, stream>>>(
        Wu + (size_t)l * 128 * GN_H, packWu + (size_t)l * 128 * GN_H, 128);
  }
  pack_w_kernel<<<(64 * 64 + 255) / 256, 256, 0, stream>>>(W1, packW1, 64);

  const float* hprev = x;
  for (int l = 0; l < GN_L; ++l) {
    float* ubuf = (l & 1) ? hB : hA;
    hipMemsetAsync(agg, 0, (size_t)GN_N * GN_H * 4, stream);
    hipMemsetAsync(stats, 0, 128 * 4, stream);
    msg_kernel<<<(GN_E / 16 + 7) / 8, 256, 0, stream>>>(
        hprev, srcp, dstp, packWm + (size_t)l * 128 * GN_H, bm + l * GN_H,
        invdeg, agg);
    update_kernel<<<(GN_N / 16 + 7) / 8, 256, 0, stream>>>(
        hprev, agg, packWu + (size_t)l * 128 * GN_H, bu + l * GN_H, a, l, ubuf,
        stats);
    bn_kernel<<<(GN_N * GN_H + 255) / 256, 256, 0, stream>>>(
        ubuf, stats, gamma + l * GN_H, beta + l * GN_H);
    hprev = ubuf;
  }

  // global mean pool
  hipMemsetAsync(gpool, 0, (size_t)GN_G * GN_H * 4, stream);
  hipMemsetAsync(gcnt, 0, (size_t)GN_G * 4, stream);
  cnt_kernel<<<(GN_N + 255) / 256, 256, 0, stream>>>(batch, gcnt);
  pool_kernel<<<(GN_N * GN_H + 255) / 256, 256, 0, stream>>>(hprev, batch, gpool);
  gdiv_kernel<<<(GN_G * GN_H + 255) / 256, 256, 0, stream>>>(gpool, gcnt);

  // head
  head_kernel<<<((GN_G + 15) / 16 + 7) / 8, 256, 0, stream>>>(gpool, packW1, b1, t1);
  head_out_kernel<<<(GN_G * GN_OUT + 255) / 256, 256, 0, stream>>>(
      t1, W2, b2, (float*)d_out);
}
